// NodeAssembly_14757507629490
// MI455X (gfx1250) — matrix operand order repaired and checked by reading, not compiled, not run on hardware
//
#include <hip/hip_runtime.h>
#include <hip/hip_bf16.h>
#include <stdint.h>

typedef __attribute__((ext_vector_type(16))) _Float16 v16h;
typedef __attribute__((ext_vector_type(8)))  float    v8f;

#if defined(__has_builtin)
#if __has_builtin(__builtin_amdgcn_tensor_load_to_lds)
#define HAS_TDM 1
#endif
#endif

#ifdef HAS_TDM
typedef __attribute__((ext_vector_type(4))) unsigned int u32x4;
typedef __attribute__((ext_vector_type(8))) int          i32x8;
typedef __attribute__((ext_vector_type(4))) int          i32x4;

// Issue one TDM tensor_load_to_lds of a rows x cols f32 tile (row stride
// strideElems) from gsrc into LDS byte offset ldsOff. D# layout per
// CDNA5 ISA (group0: count/lds/global/type,
// group1: data_size=4B, tensor dims, tile dims, dim0 stride; groups 2/3
// zero => 2-D tensor). Tracked by TENSORcnt.
__device__ __forceinline__ void tdm_load_tile_f32(const float* gsrc, unsigned ldsOff,
                                                  int rows, int cols, int strideElems) {
  unsigned long long ga = (unsigned long long)(uintptr_t)gsrc;
  u32x4 g0;
  g0[0] = 1u;                                            // count=1, user D#
  g0[1] = ldsOff;                                        // lds_addr (bytes)
  g0[2] = (unsigned)(ga & 0xFFFFFFFFull);                // global_addr[31:0]
  g0[3] = (unsigned)((ga >> 32) & 0x01FFFFFFull)         // global_addr[56:32]
          | (2u << 30);                                  // type = 2 ("image")
  i32x8 g1;
  g1[0] = 0x20000;                                       // data_size = 2 (4 bytes)
  g1[1] = (int)(((unsigned)cols & 0xFFFFu) << 16);       // tensor_dim0[15:0]
  g1[2] = (int)((((unsigned)cols >> 16) & 0xFFFFu)       // tensor_dim0[31:16]
          | (((unsigned)rows & 0xFFFFu) << 16));         // tensor_dim1[15:0]
  g1[3] = (int)((((unsigned)rows >> 16) & 0xFFFFu)       // tensor_dim1[31:16]
          | (((unsigned)cols & 0xFFFFu) << 16));         // tile_dim0 = cols
  g1[4] = (int)((unsigned)rows & 0xFFFFu);               // tile_dim1 = rows
  g1[5] = strideElems;                                   // tensor_dim0_stride[31:0]
  g1[6] = 0;                                             // stride[47:32], dim1_stride lo
  g1[7] = 0;
  i32x4 z4 = {0, 0, 0, 0};
#if __clang_major__ >= 23
  i32x8 z8 = {0, 0, 0, 0, 0, 0, 0, 0};
  __builtin_amdgcn_tensor_load_to_lds(g0, g1, z4, z4, z8, 0);
#else
  __builtin_amdgcn_tensor_load_to_lds(g0, g1, z4, z4, 0);
#endif
}
#endif // HAS_TDM

// ---------- order-preserving float <-> uint key ----------
__device__ __forceinline__ unsigned fkeyU(float f) {
  unsigned u = __float_as_uint(f);
  return (u & 0x80000000u) ? ~u : (u | 0x80000000u);
}
__device__ __forceinline__ float fkeyInv(unsigned k) {
  unsigned u = (k & 0x80000000u) ? (k & 0x7FFFFFFFu) : ~k;
  return __uint_as_float(u);
}

// ---------- phase 0: init node-side scratch ----------
__global__ void init_node(unsigned* segmax, float* denom, int* rem, int* clus,
                          float* sc, int N) {
  int i = blockIdx.x * blockDim.x + threadIdx.x;
  if (i <= N) { rem[i] = 1; clus[i] = 0; sc[i] = 1.0f; }
  if (i < N)  { segmax[i] = 0u; denom[i] = 0.0f; }
}

__global__ void zero_f32(float* p, long total) {
  long i = (long)blockIdx.x * blockDim.x + threadIdx.x;
  if (i < total) p[i] = 0.0f;
}

// ---------- phase 1: dual projection GEMV via TDM -> LDS -> WMMA ----------
// Each wave owns one 16-row tile of x. The TDM streams 16x32 f32 sub-tiles
// into double-buffered LDS (overlapped with compute, TENSORcnt-paced); the
// wave converts to f16 fragments and accumulates with
// v_wmma_f32_16x16x32_f16. B packs W[:C] in column 0 and W[C:] in column 1.
// Accumulator column 0 -> a[], column 1 -> d[].
__global__ void gemv_wmma(const float* __restrict__ x, const float* __restrict__ W,
                          float* __restrict__ aOut, float* __restrict__ dOut,
                          int N, int C) {
  __shared__ _Float16 Wh[1024];            // 2*C halves (C <= 512)
  __shared__ float    tiles[8 * 2 * 512];  // 8 waves x 2 buffers x (16x32) f32
  int tid = threadIdx.x;
  int limit = 2 * C; if (limit > 1024) limit = 1024;
  for (int i = tid; i < limit; i += blockDim.x) Wh[i] = (_Float16)W[i];
  __syncthreads();

  int lane = tid & 31;
  int wave = tid >> 5;
  int tile = blockIdx.x * (blockDim.x >> 5) + wave;
  int row0 = tile * 16;
  if (row0 >= N) return;                   // wave-uniform exit, EXEC stays full

  int rloc  = lane & 15;
  int hi    = lane >> 4;                   // 0: lanes 0-15, 1: lanes 16-31
  int base8 = hi * 8;
  int col   = lane & 15;
  v8f c = {};

#ifdef HAS_TDM
  // N % 16 == 0 for this problem, so every TDM tile is full.
  const float* xbase = x + (size_t)row0 * C;
  float* buf0 = &tiles[(wave * 2 + 0) * 512];
  float* buf1 = &tiles[(wave * 2 + 1) * 512];
  unsigned off0 = (unsigned)(uintptr_t)buf0;
  unsigned off1 = (unsigned)(uintptr_t)buf1;

  tdm_load_tile_f32(xbase, off0, 16, 32, C);           // prologue: tile k0=0
  int buf = 0;
  for (int k0 = 0; k0 < C; k0 += 32, buf ^= 1) {
    int nxt = k0 + 32;
    if (nxt < C) {                                     // pipeline next tile
      tdm_load_tile_f32(xbase + nxt, buf ? off0 : off1, 16, 32, C);
      __builtin_amdgcn_s_wait_tensorcnt(1);            // older TDM complete
    } else {
      __builtin_amdgcn_s_wait_tensorcnt(0);            // drain
    }
    const float* tp = buf ? buf1 : buf0;               // 16x32 row-major tile
    v16h a;
#pragma unroll
    for (int j = 0; j < 8; ++j) a[j]     = (_Float16)tp[rloc * 32 + base8 + j];
#pragma unroll
    for (int j = 0; j < 8; ++j) a[8 + j] = (_Float16)tp[rloc * 32 + 16 + base8 + j];
    v16h b;
    int kb = k0 + hi * 16;
#pragma unroll
    for (int j = 0; j < 16; ++j) {
      _Float16 v = (_Float16)0.0f;
      if (col == 0)      v = Wh[k0 + (j < 8 ? base8 + j : 16 + base8 + (j - 8))];
      else if (col == 1) v = Wh[C + k0 + (j < 8 ? base8 + j : 16 + base8 + (j - 8))];
      b[j] = v;
    }
    c = __builtin_amdgcn_wmma_f32_16x16x32_f16(false, a, false, b,
                                               (short)0, c, false, false);
  }
#else
  // Fallback: direct global loads of the A fragment.
  int r = row0 + rloc; if (r >= N) r = N - 1;
  const float* xr = x + (size_t)r * C;
  for (int k0 = 0; k0 < C; k0 += 32) {
    __builtin_prefetch(xr + k0 + 64, 0, 1);
    v16h a;
#pragma unroll
    for (int j = 0; j < 8; ++j) a[j]     = (_Float16)xr[k0 + base8 + j];
#pragma unroll
    for (int j = 0; j < 8; ++j) a[8 + j] = (_Float16)xr[k0 + 16 + base8 + j];
    v16h b;
    int kb = k0 + hi * 16;
#pragma unroll
    for (int j = 0; j < 16; ++j) {
      _Float16 v = (_Float16)0.0f;
      if (col == 0)      v = Wh[k0 + (j < 8 ? base8 + j : 16 + base8 + (j - 8))];
      else if (col == 1) v = Wh[C + k0 + (j < 8 ? base8 + j : 16 + base8 + (j - 8))];
      b[j] = v;
    }
    c = __builtin_amdgcn_wmma_f32_16x16x32_f16(false, a, false, b,
                                               (short)0, c, false, false);
  }
#endif

  // C/D layout: VGPR m, lanes 0-15 -> M=m, N=lane ; lanes 16-31 -> M=8+m
  if (col < 2) {
    float* out = (col == 0) ? aOut : dOut;
    int mbase = hi * 8;
#pragma unroll
    for (int m = 0; m < 8; ++m) {
      int rr = row0 + mbase + m;
      if (rr < N) out[rr] = c[m];
    }
  }
}

// ---------- phase 2: edge softmax over destination segments ----------
__global__ void edge_score(const int* __restrict__ src, const int* __restrict__ dst,
                           const float* __restrict__ a, const float* __restrict__ d,
                           const float* __restrict__ bp, float* __restrict__ e,
                           unsigned* __restrict__ segmax, int E) {
  int i = blockIdx.x * blockDim.x + threadIdx.x;
  if (i >= E) return;
  float v = a[src[i]] + d[dst[i]] + bp[0];
  e[i] = v;
  atomicMax(&segmax[dst[i]], fkeyU(v));
}

__global__ void edge_exp(const int* __restrict__ dst, float* __restrict__ e,
                         const unsigned* __restrict__ segmax,
                         float* __restrict__ denom, int E) {
  int i = blockIdx.x * blockDim.x + threadIdx.x;
  if (i >= E) return;
  float m  = fkeyInv(segmax[dst[i]]);
  float ee = __expf(e[i] - m);
  e[i] = ee;
  atomicAdd(&denom[dst[i]], ee);
}

__global__ void edge_norm(const int* __restrict__ dst, float* __restrict__ e,
                          const float* __restrict__ denom,
                          unsigned long long* __restrict__ kv, int E, int NP) {
  int i = blockIdx.x * blockDim.x + threadIdx.x;
  if (i >= NP) return;
  if (i < E) {
    float v = e[i] / (denom[dst[i]] + 1e-16f);
    e[i] = v;
    unsigned key = ~fkeyU(v);   // ascending sort of key == descending v
    kv[i] = ((unsigned long long)key << 32) | (unsigned)i;
  } else {
    kv[i] = ~0ull;              // pads sort to the end
  }
}

// ---------- bitonic sort step on packed u64 (key<<32 | payload) ----------
__global__ void bitonic_step(unsigned long long* __restrict__ kv, int j, int k, int n) {
  int i = blockIdx.x * blockDim.x + threadIdx.x;
  if (i >= n) return;
  int ixj = i ^ j;
  if (ixj > i) {
    unsigned long long A = kv[i], B = kv[ixj];
    bool asc = ((i & k) == 0);
    if ((asc && A > B) || (!asc && A < B)) { kv[i] = B; kv[ixj] = A; }
  }
}

// ---------- phase 3: sequential greedy matching (inherently serial) ----------
__global__ void greedy_match(const unsigned long long* __restrict__ kv,
                             const int* __restrict__ src, const int* __restrict__ dst,
                             const float* __restrict__ e,
                             int* __restrict__ rem, int* __restrict__ clus,
                             float* __restrict__ sc, int E, int N) {
  if (threadIdx.x != 0 || blockIdx.x != 0) return;
  int cnt = 0;
  for (int i = 0; i < E; ++i) {
    unsigned idx = (unsigned)(kv[i] & 0xFFFFFFFFu);
    int s = src[idx], t = dst[idx];
    if (rem[s] && rem[t]) {
      clus[s] = cnt; clus[t] = cnt;
      rem[s] = 0;    rem[t] = 0;
      sc[cnt] = e[idx];
      ++cnt;
    }
  }
  int r = 0;
  for (int n = 0; n < N; ++n)
    if (rem[n]) { clus[n] = cnt + r; ++r; }
}

// ---------- phase 4: cluster feature aggregation ----------
__global__ void aggregate(const float* __restrict__ x, const int* __restrict__ clus,
                          float* __restrict__ newx, int N, int C) {
  long i = (long)blockIdx.x * blockDim.x + threadIdx.x;
  long total = (long)N * C;
  if (i >= total) return;
  int n = (int)(i / C), c = (int)(i % C);
  atomicAdd(&newx[(long)clus[n] * C + c], x[i]);
}

__global__ void scale_newx(float* __restrict__ newx, const float* __restrict__ sc,
                           int N, int C) {
  long i = (long)blockIdx.x * blockDim.x + threadIdx.x;
  long total = (long)N * C;
  if (i >= total) return;
  newx[i] *= sc[i / C];
}

// ---------- phase 5: edge remap, sort, dedup ----------
__global__ void remap_edges(const int* __restrict__ src, const int* __restrict__ dst,
                            const int* __restrict__ clus,
                            unsigned long long* __restrict__ kv, int E, int NP) {
  int i = blockIdx.x * blockDim.x + threadIdx.x;
  if (i >= NP) return;
  if (i < E) {
    unsigned row = (unsigned)clus[src[i]];
    unsigned col = (unsigned)clus[dst[i]];
    unsigned key = (row << 16) | col;      // N < 2^16, both fit
    kv[i] = ((unsigned long long)key << 32) | (unsigned)i;
  } else {
    kv[i] = ~0ull;
  }
}

__global__ void finalize_edges(const unsigned long long* __restrict__ kv,
                               int* __restrict__ outE, int E) {
  int i = blockIdx.x * blockDim.x + threadIdx.x;
  if (i >= E) return;
  unsigned key = (unsigned)(kv[i] >> 32);
  int row = (int)(key >> 16), col = (int)(key & 0xFFFFu);
  bool dup = (i > 0) && ((unsigned)(kv[i - 1] >> 32) == key);
  outE[i]     = dup ? -1 : row;
  outE[E + i] = dup ? -1 : col;
}

// ---------- host orchestration ----------
extern "C" void kernel_launch(void* const* d_in, const int* in_sizes, int n_in,
                              void* d_out, int out_size, void* d_ws, size_t ws_size,
                              hipStream_t stream) {
  const float* x    = (const float*)d_in[0];
  const int*   eidx = (const int*)d_in[1];
  const float* W    = (const float*)d_in[2];
  const float* bp   = (const float*)d_in[3];

  int C = in_sizes[2] / 2;
  int N = in_sizes[0] / C;
  int E = in_sizes[1] / 2;
  const int* src = eidx;
  const int* dst = eidx + E;

  int NP = 1; while (NP < E) NP <<= 1;   // pad edges to power of two

  // carve workspace (d_ws is 8-byte aligned from hipMalloc)
  char* w = (char*)d_ws;
  unsigned long long* kv1 = (unsigned long long*)w; w += (size_t)NP * 8;
  unsigned long long* kv2 = (unsigned long long*)w; w += (size_t)NP * 8;
  float*    aV     = (float*)w;    w += (size_t)N * 4;
  float*    dV     = (float*)w;    w += (size_t)N * 4;
  unsigned* segmax = (unsigned*)w; w += (size_t)N * 4;
  float*    denom  = (float*)w;    w += (size_t)N * 4;
  float*    eArr   = (float*)w;    w += (size_t)E * 4;
  int*      rem    = (int*)w;      w += (size_t)(N + 1) * 4;
  int*      clus   = (int*)w;      w += (size_t)(N + 1) * 4;
  float*    sc     = (float*)w;    w += (size_t)(N + 1) * 4;

  float* newx  = (float*)d_out;                  // N*C floats
  int*   outE  = (int*)((float*)d_out + (size_t)N * C);  // 2*E int32 (bit-cast region)

  const int TB = 256;
  long totalNC = (long)N * C;

  // init
  init_node<<<(N + 1 + TB - 1) / TB, TB, 0, stream>>>(segmax, denom, rem, clus, sc, N);
  zero_f32<<<(int)((totalNC + TB - 1) / TB), TB, 0, stream>>>(newx, totalNC);

  // phase 1: TDM + WMMA dual projection (8 waves / block, 1 tile / wave)
  {
    int tiles  = (N + 15) / 16;
    int blocks = (tiles + 7) / 8;
    gemv_wmma<<<blocks, 256, 0, stream>>>(x, W, aV, dV, N, C);
  }

  // phase 2: edge softmax
  int eb = (E + TB - 1) / TB;
  int pb = (NP + TB - 1) / TB;
  edge_score<<<eb, TB, 0, stream>>>(src, dst, aV, dV, bp, eArr, segmax, E);
  edge_exp  <<<eb, TB, 0, stream>>>(dst, eArr, segmax, denom, E);
  edge_norm <<<pb, TB, 0, stream>>>(dst, eArr, denom, kv1, E, NP);

  // phase 3a: bitonic sort by descending attention
  for (int k = 2; k <= NP; k <<= 1)
    for (int j = k >> 1; j > 0; j >>= 1)
      bitonic_step<<<pb, TB, 0, stream>>>(kv1, j, k, NP);

  // phase 3b: sequential greedy matching + rank assignment
  greedy_match<<<1, 32, 0, stream>>>(kv1, src, dst, eArr, rem, clus, sc, E, N);

  // phase 4: aggregate + scale
  aggregate <<<(int)((totalNC + TB - 1) / TB), TB, 0, stream>>>(x, clus, newx, N, C);
  scale_newx<<<(int)((totalNC + TB - 1) / TB), TB, 0, stream>>>(newx, sc, N, C);

  // phase 5: remap, lexsort, dedup
  remap_edges<<<pb, TB, 0, stream>>>(src, dst, clus, kv2, E, NP);
  for (int k = 2; k <= NP; k <<= 1)
    for (int j = k >> 1; j > 0; j >>= 1)
      bitonic_step<<<pb, TB, 0, stream>>>(kv2, j, k, NP);
  finalize_edges<<<eb, TB, 0, stream>>>(kv2, outE, E);
}
